// MultiHeadAttention_75368086110671
// MI455X (gfx1250) — compile-verified, matrix operand order checked by reading, not run on hardware
//
#include <hip/hip_runtime.h>
#include <hip/hip_bf16.h>

typedef __attribute__((ext_vector_type(16))) _Float16 v16h;
typedef __attribute__((ext_vector_type(8)))  float    v8f;

#define S_LEN 2048
#define NHEAD 16
#define DK 64
#define DM 1024
#define NEGINF (-1.0e9f)

// ---------------------------------------------------------------------------
// Fragment helpers (CDNA5 wave32 WMMA 16x16x32 f16 layouts, ISA 7.12.2):
// A (16xK): lane L(0..15) = row M=L, halves 0..7 -> K=kb+off+0..7,
//           halves 8..15 -> K=kb+off+16..23, off = (lane>=16) ? 8 : 0.
// B (Kx16) mirrors A with lane = output column n; since y = X*W^T, lane n
// reads W[n, k...] contiguously.
// C/D (16x16 f32): v8f elem r -> M = r + 8*(lane>=16), N = lane&15.
// ---------------------------------------------------------------------------

__device__ __forceinline__ v16h ldfrag_f16(const _Float16* __restrict__ p,
                                           size_t ld, int row, int kb) {
  int lane = threadIdx.x & 31;
  int r    = lane & 15;
  int off  = (lane & 16) ? 8 : 0;
  const _Float16* q = p + (size_t)(row + r) * ld + kb + off;
  v16h a;
#pragma unroll
  for (int i = 0; i < 8; ++i) { a[i] = q[i]; a[i + 8] = q[i + 16]; }
  return a;
}

__device__ __forceinline__ v16h ldfrag_f32(const float* __restrict__ p,
                                           size_t ld, int row, int kb) {
  int lane = threadIdx.x & 31;
  int r    = lane & 15;
  int off  = (lane & 16) ? 8 : 0;
  const float* q = p + (size_t)(row + r) * ld + kb + off;
  v16h a;
#pragma unroll
  for (int i = 0; i < 8; ++i) {
    a[i]     = (_Float16)q[i];
    a[i + 8] = (_Float16)q[i + 16];
  }
  return a;
}

__device__ __forceinline__ v8f wmma16(v16h a, v16h b, v8f c) {
  return __builtin_amdgcn_wmma_f32_16x16x32_f16(false, a, false, b,
                                                (short)0, c, false, false);
}

__device__ __forceinline__ void st_tile_f16(_Float16* dst, size_t ld,
                                            v8f c, int mo, int n) {
#pragma unroll
  for (int r = 0; r < 8; ++r)
    dst[(size_t)(r + mo) * ld + n] = (_Float16)c[r];
}

// ---------------------------------------------------------------------------
// Kernel 1: projection GEMM  out = X @ W^T  (f32 in, f16 out, f32 accum)
// X: [4096,1024], W: [1024,1024] row-major (torch Linear weight).
// mode 0: out[((b*16+h)*2048+s)*64 + d]   (q16 / k16, head-major)
// mode 1: out[((b*16+h)*64+d)*2048 + s]   (vT16, head-transposed)
// One wave -> 16x64 tile (one head slice). 8 waves/block.
// ---------------------------------------------------------------------------
__global__ __launch_bounds__(256) void proj_kernel(
    const float* __restrict__ X, const float* __restrict__ W,
    _Float16* __restrict__ out, int mode) {
  int wave = threadIdx.x >> 5;
  int gid  = blockIdx.x * 8 + wave;     // 4096 waves total
  int mt   = gid >> 4;                  // 256 row tiles of 16
  int h    = gid & 15;                  // 16 column tiles of 64 == head id
  int lane = threadIdx.x & 31;
  int n    = lane & 15;
  int mo   = (lane & 16) ? 8 : 0;
  int mrow = mt * 16;

  v8f o0 = {}, o1 = {}, o2 = {}, o3 = {};
  for (int kb = 0; kb < DM; kb += 32) {
    v16h a  = ldfrag_f32(X, DM, mrow, kb);
    v16h b0 = ldfrag_f32(W, DM, h * 64 + 0,  kb);
    v16h b1 = ldfrag_f32(W, DM, h * 64 + 16, kb);
    v16h b2 = ldfrag_f32(W, DM, h * 64 + 32, kb);
    v16h b3 = ldfrag_f32(W, DM, h * 64 + 48, kb);
    o0 = wmma16(a, b0, o0);
    o1 = wmma16(a, b1, o1);
    o2 = wmma16(a, b2, o2);
    o3 = wmma16(a, b3, o3);
  }

  int bi = mrow >> 11;                  // batch
  int s0 = mrow & (S_LEN - 1);          // row within batch
  if (mode == 0) {
    _Float16* dst = out + ((size_t)(bi * NHEAD + h) * S_LEN + s0) * DK;
    st_tile_f16(dst + 0,  DK, o0, mo, n);
    st_tile_f16(dst + 16, DK, o1, mo, n);
    st_tile_f16(dst + 32, DK, o2, mo, n);
    st_tile_f16(dst + 48, DK, o3, mo, n);
  } else {
    _Float16* dst = out + (size_t)(bi * NHEAD + h) * DK * S_LEN;
    v8f oc[4] = {o0, o1, o2, o3};
#pragma unroll
    for (int t = 0; t < 4; ++t)
#pragma unroll
      for (int r = 0; r < 8; ++r)
        dst[(size_t)(t * 16 + n) * S_LEN + s0 + r + mo] = (_Float16)oc[t][r];
  }
}

// ---------------------------------------------------------------------------
// Kernel 2: attention. One wave per (b, h, 16-query tile).
// 3 sweeps over key tiles (recompute scores instead of spilling 256MB of P):
//   sweep1 rowmax, sweep2 rowsum of exp, sweep3 normalize -> write f32
//   attn_weights to d_out, stage f16 P in a 1KB LDS tile, and accumulate
//   P @ V via WMMA (V pre-transposed so B-fragments are contiguous loads).
// ---------------------------------------------------------------------------
__global__ __launch_bounds__(32) void attn_kernel(
    const _Float16* __restrict__ q16, const _Float16* __restrict__ k16,
    const _Float16* __restrict__ vT16, float* __restrict__ attnW,
    _Float16* __restrict__ attn16) {
  __shared__ _Float16 ptile[16 * 32];

  int gid  = blockIdx.x;
  int qt   = gid & 127;
  int h    = (gid >> 7) & 15;
  int bi   = gid >> 11;
  int lane = threadIdx.x & 31;
  int n    = lane & 15;
  int mo   = (lane & 16) ? 8 : 0;
  int qrow = qt * 16;

  const _Float16* qh = q16 + (size_t)(bi * NHEAD + h) * S_LEN * DK;
  const _Float16* kh = k16 + (size_t)(bi * NHEAD + h) * S_LEN * DK;
  const _Float16* vh = vT16 + (size_t)(bi * NHEAD + h) * DK * S_LEN;
  float* aw = attnW + (size_t)(bi * NHEAD + h) * S_LEN * S_LEN;

  v16h aq0 = ldfrag_f16(qh, DK, qrow, 0);
  v16h aq1 = ldfrag_f16(qh, DK, qrow, 32);

  // ---- sweep 1: row max (exact softmax, matching reference NEG mask) ----
  float rmax[8];
#pragma unroll
  for (int r = 0; r < 8; ++r) rmax[r] = -3.0e38f;
  for (int jt = 0; jt <= qt; ++jt) {
    v8f c = {};
    c = wmma16(aq0, ldfrag_f16(kh, DK, jt * 16, 0),  c);
    c = wmma16(aq1, ldfrag_f16(kh, DK, jt * 16, 32), c);
#pragma unroll
    for (int r = 0; r < 8; ++r) {
      int m = qrow + r + mo, j = jt * 16 + n;
      float s = (j <= m) ? c[r] * 0.125f : NEGINF;
      rmax[r] = fmaxf(rmax[r], s);
    }
  }
#pragma unroll
  for (int r = 0; r < 8; ++r)
    for (int w = 1; w < 16; w <<= 1)
      rmax[r] = fmaxf(rmax[r], __shfl_xor(rmax[r], w, 32));

  // ---- sweep 2: row sum of exp ----
  float rsum[8];
#pragma unroll
  for (int r = 0; r < 8; ++r) rsum[r] = 0.f;
  for (int jt = 0; jt <= qt; ++jt) {
    v8f c = {};
    c = wmma16(aq0, ldfrag_f16(kh, DK, jt * 16, 0),  c);
    c = wmma16(aq1, ldfrag_f16(kh, DK, jt * 16, 32), c);
#pragma unroll
    for (int r = 0; r < 8; ++r) {
      int m = qrow + r + mo, j = jt * 16 + n;
      float s = (j <= m) ? c[r] * 0.125f : NEGINF;
      rsum[r] += __expf(s - rmax[r]);
    }
  }
#pragma unroll
  for (int r = 0; r < 8; ++r) {
    for (int w = 1; w < 16; w <<= 1)
      rsum[r] += __shfl_xor(rsum[r], w, 32);
    rsum[r] = 1.0f / rsum[r];
  }

  // ---- sweep 3: normalize, emit attn_weights, accumulate P @ V ----
  v8f o0 = {}, o1 = {}, o2 = {}, o3 = {};
  int npair = (qt + 2) >> 1;            // pairs of key tiles covering 0..qt
  for (int jp = 0; jp < npair; ++jp) {
#pragma unroll
    for (int half = 0; half < 2; ++half) {
      int jt = jp * 2 + half;
      v8f p = {};
      if (jt <= qt) {                   // wave-uniform branch
        v8f c = {};
        c = wmma16(aq0, ldfrag_f16(kh, DK, jt * 16, 0),  c);
        c = wmma16(aq1, ldfrag_f16(kh, DK, jt * 16, 32), c);
#pragma unroll
        for (int r = 0; r < 8; ++r) {
          int m = qrow + r + mo, j = jt * 16 + n;
          float s = (j <= m) ? c[r] * 0.125f : NEGINF;
          p[r] = __expf(s - rmax[r]) * rsum[r];   // masked -> exactly 0
        }
      }
#pragma unroll
      for (int r = 0; r < 8; ++r) {
        aw[(size_t)(qrow + r + mo) * S_LEN + jt * 16 + n] = p[r];
        ptile[(r + mo) * 32 + half * 16 + n] = (_Float16)p[r];
      }
    }
    __syncthreads();
    v16h pA = ldfrag_f16(ptile, 32, 0, 0);      // C-frag -> A-frag via LDS
    __syncthreads();
    int jb = jp * 32;
    o0 = wmma16(pA, ldfrag_f16(vh, S_LEN, 0,  jb), o0);
    o1 = wmma16(pA, ldfrag_f16(vh, S_LEN, 16, jb), o1);
    o2 = wmma16(pA, ldfrag_f16(vh, S_LEN, 32, jb), o2);
    o3 = wmma16(pA, ldfrag_f16(vh, S_LEN, 48, jb), o3);
  }

  // zero-fill fully-masked future key tiles (d_out is poisoned)
  for (int j = npair * 32; j < S_LEN; j += 16)
#pragma unroll
    for (int r = 0; r < 8; ++r)
      aw[(size_t)(qrow + r + mo) * S_LEN + j + n] = 0.f;

  // attn_out tile -> f16 workspace in [B, S, D_MODEL] layout
  _Float16* ao = attn16 + ((size_t)bi * S_LEN + qrow) * DM + h * DK;
  st_tile_f16(ao + 0,  DM, o0, mo, n);
  st_tile_f16(ao + 16, DM, o1, mo, n);
  st_tile_f16(ao + 32, DM, o2, mo, n);
  st_tile_f16(ao + 48, DM, o3, mo, n);
}

// ---------------------------------------------------------------------------
// Kernel 3: output projection + residual:  y = attn16 @ W_O^T + Q
// ---------------------------------------------------------------------------
__global__ __launch_bounds__(256) void outproj_kernel(
    const _Float16* __restrict__ A, const float* __restrict__ WO,
    const float* __restrict__ resid, float* __restrict__ y) {
  int wave = threadIdx.x >> 5;
  int gid  = blockIdx.x * 8 + wave;
  int mt   = gid >> 4;
  int nt   = gid & 15;
  int lane = threadIdx.x & 31;
  int n    = lane & 15;
  int mo   = (lane & 16) ? 8 : 0;
  int mrow = mt * 16;

  v8f o0 = {}, o1 = {}, o2 = {}, o3 = {};
  for (int kb = 0; kb < DM; kb += 32) {
    v16h a = ldfrag_f16(A, DM, mrow, kb);
    o0 = wmma16(a, ldfrag_f32(WO, DM, nt * 64 + 0,  kb), o0);
    o1 = wmma16(a, ldfrag_f32(WO, DM, nt * 64 + 16, kb), o1);
    o2 = wmma16(a, ldfrag_f32(WO, DM, nt * 64 + 32, kb), o2);
    o3 = wmma16(a, ldfrag_f32(WO, DM, nt * 64 + 48, kb), o3);
  }
  v8f oc[4] = {o0, o1, o2, o3};
#pragma unroll
  for (int t = 0; t < 4; ++t)
#pragma unroll
    for (int r = 0; r < 8; ++r) {
      size_t idx = (size_t)(mrow + r + mo) * DM + nt * 64 + t * 16 + n;
      y[idx] = oc[t][r] + resid[idx];
    }
}

// ---------------------------------------------------------------------------
// Kernel 4: LayerNorm over last dim (1024). One 256-thread block per row.
// ---------------------------------------------------------------------------
__global__ __launch_bounds__(256) void ln_kernel(
    const float* __restrict__ y, const float* __restrict__ gamma,
    const float* __restrict__ beta, float* __restrict__ out) {
  __shared__ float red[8];
  int row = blockIdx.x;
  const float* x = y + (size_t)row * DM;

  float s = 0.f;
  for (int i = threadIdx.x; i < DM; i += 256) s += x[i];
  for (int w = 1; w < 32; w <<= 1) s += __shfl_xor(s, w, 32);
  if ((threadIdx.x & 31) == 0) red[threadIdx.x >> 5] = s;
  __syncthreads();
  float tot = 0.f;
#pragma unroll
  for (int i = 0; i < 8; ++i) tot += red[i];
  float mean = tot * (1.0f / DM);
  __syncthreads();

  float v = 0.f;
  for (int i = threadIdx.x; i < DM; i += 256) {
    float d = x[i] - mean;
    v += d * d;
  }
  for (int w = 1; w < 32; w <<= 1) v += __shfl_xor(v, w, 32);
  if ((threadIdx.x & 31) == 0) red[threadIdx.x >> 5] = v;
  __syncthreads();
  float vt = 0.f;
#pragma unroll
  for (int i = 0; i < 8; ++i) vt += red[i];
  float rstd = rsqrtf(vt * (1.0f / DM) + 1e-5f);

  for (int i = threadIdx.x; i < DM; i += 256)
    out[(size_t)row * DM + i] = (x[i] - mean) * rstd * gamma[i] + beta[i];
}

// ---------------------------------------------------------------------------
extern "C" void kernel_launch(void* const* d_in, const int* in_sizes, int n_in,
                              void* d_out, int out_size, void* d_ws,
                              size_t ws_size, hipStream_t stream) {
  (void)in_sizes; (void)n_in; (void)out_size; (void)ws_size;
  const float* Q    = (const float*)d_in[0];
  const float* K    = (const float*)d_in[1];
  const float* V    = (const float*)d_in[2];
  // d_in[3] = mask (causal, known statically) -- unused
  const float* W_Q  = (const float*)d_in[4];
  const float* W_K  = (const float*)d_in[5];
  const float* W_V  = (const float*)d_in[6];
  const float* W_O  = (const float*)d_in[7];
  const float* ln_g = (const float*)d_in[8];
  const float* ln_b = (const float*)d_in[9];

  const size_t proj_elems = (size_t)2 * NHEAD * S_LEN * DK;  // 4,194,304
  _Float16* q16    = (_Float16*)d_ws;
  _Float16* k16    = q16    + proj_elems;
  _Float16* vT16   = k16    + proj_elems;
  _Float16* attn16 = vT16   + proj_elems;
  float*    yws    = (float*)(attn16 + proj_elems);

  float* out_f   = (float*)d_out;                         // [2,2048,1024]
  float* attnW   = out_f + (size_t)2 * S_LEN * DM;        // [2,16,2048,2048]

  proj_kernel<<<512, 256, 0, stream>>>(Q, W_Q, q16, 0);
  proj_kernel<<<512, 256, 0, stream>>>(K, W_K, k16, 0);
  proj_kernel<<<512, 256, 0, stream>>>(V, W_V, vT16, 1);
  attn_kernel<<<2 * NHEAD * (S_LEN / 16), 32, 0, stream>>>(q16, k16, vT16,
                                                           attnW, attn16);
  outproj_kernel<<<512, 256, 0, stream>>>(attn16, W_O, Q, yws);
  ln_kernel<<<2 * S_LEN, 256, 0, stream>>>(yws, ln_g, ln_b, out_f);
}
